// Backbone_Net_77936476553268
// MI455X (gfx1250) — matrix operand order repaired and checked by reading, not compiled, not run on hardware
//
#include <hip/hip_runtime.h>

typedef __attribute__((ext_vector_type(16))) _Float16 v16h;
typedef __attribute__((ext_vector_type(8)))  float    v8f;

__device__ __forceinline__ v8f wmma_f16(v16h a, v16h b, v8f c) {
  // D = A(16x32 f16) x B(32x16 f16) + C(16x16 f32)
  return __builtin_amdgcn_wmma_f32_16x16x32_f16(false, a, false, b, (short)0, c, false, false);
}

// ---------------- prep kernels ----------------

__global__ void bn_fold_kernel(const float* __restrict__ g, const float* __restrict__ b,
                               const float* __restrict__ m, const float* __restrict__ v,
                               float* __restrict__ scale, float* __restrict__ bias, int C) {
  int i = blockIdx.x * 256 + threadIdx.x;
  if (i < C) {
    float s = g[i] * rsqrtf(v[i] + 1e-5f);
    scale[i] = s;
    bias[i]  = b[i] - m[i] * s;
  }
}

// OIHW fp32 -> [O][Kpad] f16 with k = (kh*KW+kw)*Cin + ci, zero pad to Kpad
__global__ void prep_w_kernel(const float* __restrict__ w, _Float16* __restrict__ wt,
                              int O, int Cin, int KH, int Kpad) {
  long i = (long)blockIdx.x * 256 + threadIdx.x;
  if (i >= (long)O * Kpad) return;
  int o = (int)(i / Kpad), k = (int)(i % Kpad);
  int K = Cin * KH * KH;
  float val = 0.f;
  if (k < K) {
    int ci = k % Cin, t = k / Cin, kh = t / KH, kw = t % KH;
    val = w[(((long)o * Cin + ci) * KH + kh) * KH + kw];
  }
  wt[i] = (_Float16)val;
}

// fc_w [512][8192] with src k = c*16 + hw  ->  dst k = hw*512 + c (NHWC flatten)
__global__ void prep_fc_kernel(const float* __restrict__ w, _Float16* __restrict__ wt) {
  long i = (long)blockIdx.x * 256 + threadIdx.x;
  if (i >= (long)512 * 8192) return;
  int o = (int)(i / 8192), kd = (int)(i % 8192);
  int hw = kd / 512, c = kd % 512;
  wt[i] = (_Float16)w[(long)o * 8192 + c * 16 + hw];
}

__global__ void nchw2nhwc_kernel(const float* __restrict__ x, _Float16* __restrict__ y,
                                 int B, int C, int H, int W) {
  long i = (long)blockIdx.x * 256 + threadIdx.x;
  if (i >= (long)B * C * H * W) return;
  int c = (int)(i % C); long t = i / C;
  int w = (int)(t % W); t /= W;
  int h = (int)(t % H); int b = (int)(t / H);
  y[i] = (_Float16)x[(((long)b * C + c) * H + h) * W + w];
}

// 3x3/2 pad1 maxpool, NHWC f16
__global__ void maxpool_kernel(const _Float16* __restrict__ in, _Float16* __restrict__ out,
                               int B, int C, int IH, int OH) {
  long i = (long)blockIdx.x * 256 + threadIdx.x;
  if (i >= (long)B * OH * OH * C) return;
  int c = (int)(i % C); long t = i / C;
  int ow = (int)(t % OH); t /= OH;
  int oh = (int)(t % OH); int b = (int)(t / OH);
  float mx = -1e30f;
  for (int kh = 0; kh < 3; kh++)
    for (int kw = 0; kw < 3; kw++) {
      int ih = oh * 2 - 1 + kh, iw = ow * 2 - 1 + kw;
      if ((unsigned)ih < (unsigned)IH && (unsigned)iw < (unsigned)IH) {
        float v = (float)in[(((long)b * IH + ih) * IH + iw) * C + c];
        mx = v > mx ? v : mx;
      }
    }
  out[i] = (_Float16)mx;
}

// ---------------- WMMA implicit-GEMM ----------------
// Block = 256 thr = 8 waves; wave w owns M-subtile w (16 rows); block tile = 128(M) x 64(N).
// grid.x = M/128, grid.y = Cout/64  (all M mult of 128, all Cout mult of 64 -> no guards,
// WMMA fully unconditional, EXEC all-ones by construction).

// Epilogue: D[r + 8*(lane/16)][lane%16] per C/D VGPR layout; folded BN + residual + relu.
__device__ __forceinline__ void epilogue(const v8f* acc, long mBase, int wid, int nBase,
                                         int lcol, int lhi,
                                         const float* scale, const float* bias,
                                         const _Float16* res, _Float16* out,
                                         int Cout, int relu) {
#pragma unroll
  for (int f = 0; f < 4; f++) {
    int n = nBase + f * 16 + lcol;
    float sc = scale ? scale[n] : 1.0f;
    float bi = bias ? bias[n] : 0.0f;
#pragma unroll
    for (int r = 0; r < 8; r++) {
      long mo = mBase + wid * 16 + r + 8 * lhi;
      float v = acc[f][r] * sc + bi;
      long idx = mo * (long)Cout + n;
      if (res) v += (float)res[idx];
      if (relu && v < 0.f) v = 0.f;
      out[idx] = (_Float16)v;
    }
  }
}

// 1x1 conv / FC as GEMM. in: NHWC f16, wt: [Cout][Cin] f16 (Cin multiple of 32).
__global__ __launch_bounds__(256)
void gemm1x1_kernel(const _Float16* __restrict__ in, const _Float16* __restrict__ wt,
                    const float* __restrict__ scale, const float* __restrict__ bias,
                    const _Float16* __restrict__ res, _Float16* __restrict__ out,
                    int OW, int IW, int stride, int Cin, int Cout, int relu) {
  __shared__ __align__(32) _Float16 As[4096];  // 128x32 A tile, fragment-swizzled (8KB)
  const int tid = threadIdx.x, lane = tid & 31, wid = tid >> 5;
  const long mBase = (long)blockIdx.x * 128;
  const int nBase = blockIdx.y * 64;
  const int npix = OW * OW;

  // staging: thread -> (row sm, 16 k-values at k0); swizzle per A-fragment layout
  const int sm = tid >> 1, k0 = (tid & 1) * 16;
  long m = mBase + sm;
  int b = (int)(m / npix), rr = (int)(m % npix);
  int oh = rr / OW, ow = rr % OW;
  const _Float16* src = in + ((long)(b * IW + oh * stride) * IW + ow * stride) * Cin + k0;
  const int frag = sm >> 4, r = sm & 15, e0 = (k0 >> 4) << 3;
  _Float16* dstA = &As[frag * 512 + (r + 2 * e0) * 16];      // k0..k0+7, hi=k0/16
  _Float16* dstB = &As[frag * 512 + (r + 2 * e0) * 16 + 8];  // k0+8..k0+15, hi=k0/16

  const int lcol = lane & 15, lhi = lane >> 4;
  const _Float16* wrow = wt + (long)(nBase + lcol) * Cin + lhi * 16;
  const long wstep = (long)16 * Cin;

  v8f z = {};
  v8f acc0 = z, acc1 = z, acc2 = z, acc3 = z;

  for (int kt = 0; kt < Cin; kt += 32) {
    *(uint4*)dstA = *(const uint4*)(src + kt);
    *(uint4*)dstB = *(const uint4*)(src + kt + 8);
    __syncthreads();
    v16h a  = *(const v16h*)&As[wid * 512 + lane * 16];  // 2x ds_load_b128
    v16h b0 = *(const v16h*)(wrow + kt);
    v16h b1 = *(const v16h*)(wrow + wstep + kt);
    v16h b2 = *(const v16h*)(wrow + 2 * wstep + kt);
    v16h b3 = *(const v16h*)(wrow + 3 * wstep + kt);
    acc0 = wmma_f16(a, b0, acc0);
    acc1 = wmma_f16(a, b1, acc1);
    acc2 = wmma_f16(a, b2, acc2);
    acc3 = wmma_f16(a, b3, acc3);
    __syncthreads();
  }
  v8f accs[4] = {acc0, acc1, acc2, acc3};
  epilogue(accs, mBase, wid, nBase, lcol, lhi, scale, bias, res, out, Cout, relu);
}

// General KHxKH conv (7x7/2 and 3x3). wt: [Cout][Kpad] f16, k=(kh*KW+kw)*Cin+ci.
__global__ __launch_bounds__(256)
void convg_kernel(const _Float16* __restrict__ in, const _Float16* __restrict__ wt,
                  const float* __restrict__ scale, const float* __restrict__ bias,
                  const _Float16* __restrict__ res, _Float16* __restrict__ out,
                  int Cin, int Cout, int IH, int OH, int KH, int stride, int pad,
                  int Kpad, int relu) {
  __shared__ __align__(32) _Float16 As[4096];
  const int tid = threadIdx.x, lane = tid & 31, wid = tid >> 5;
  const long mBase = (long)blockIdx.x * 128;
  const int nBase = blockIdx.y * 64;
  const int npix = OH * OH;
  const int K = Cin * KH * KH;

  const int sm = tid >> 1, k0 = (tid & 1) * 16;
  long m = mBase + sm;
  int b = (int)(m / npix), rr = (int)(m % npix);
  int oh = rr / OH, ow = rr % OH;
  const int frag = sm >> 4, r = sm & 15, e0 = (k0 >> 4) << 3;
  _Float16* dstA = &As[frag * 512 + (r + 2 * e0) * 16];
  _Float16* dstB = &As[frag * 512 + (r + 2 * e0) * 16 + 8];

  const int lcol = lane & 15, lhi = lane >> 4;
  const _Float16* wrow = wt + (long)(nBase + lcol) * Kpad + lhi * 16;
  const long wstep = (long)16 * Kpad;

  v8f z = {};
  v8f acc0 = z, acc1 = z, acc2 = z, acc3 = z;

  for (int kt = 0; kt < Kpad; kt += 32) {
    __align__(16) _Float16 tmp[16];
#pragma unroll
    for (int j = 0; j < 16; j++) {
      int k = kt + k0 + j;
      _Float16 val = (_Float16)0.f;
      if (k < K) {
        int ci = k % Cin, t = k / Cin, kh = t / KH, kw = t % KH;
        int ih = oh * stride - pad + kh, iw = ow * stride - pad + kw;
        if ((unsigned)ih < (unsigned)IH && (unsigned)iw < (unsigned)IH)
          val = in[((long)(b * IH + ih) * IH + iw) * Cin + ci];
      }
      tmp[j] = val;
    }
    *(uint4*)dstA = *(const uint4*)tmp;
    *(uint4*)dstB = *(const uint4*)(tmp + 8);
    __syncthreads();
    v16h a  = *(const v16h*)&As[wid * 512 + lane * 16];
    v16h b0 = *(const v16h*)(wrow + kt);
    v16h b1 = *(const v16h*)(wrow + wstep + kt);
    v16h b2 = *(const v16h*)(wrow + 2 * wstep + kt);
    v16h b3 = *(const v16h*)(wrow + 3 * wstep + kt);
    acc0 = wmma_f16(a, b0, acc0);
    acc1 = wmma_f16(a, b1, acc1);
    acc2 = wmma_f16(a, b2, acc2);
    acc3 = wmma_f16(a, b3, acc3);
    __syncthreads();
  }
  v8f accs[4] = {acc0, acc1, acc2, acc3};
  epilogue(accs, mBase, wid, nBase, lcol, lhi, scale, bias, res, out, Cout, relu);
}

// Stereographic projection: one wave32 per row of [B,256] f16 -> [B,257] f32
__global__ __launch_bounds__(256)
void project_kernel(const _Float16* __restrict__ x, float* __restrict__ out, int B) {
  int lane = threadIdx.x & 31;
  int b = blockIdx.x * 8 + (threadIdx.x >> 5);
  if (b >= B) return;
  float vals[8]; float ss = 0.f;
#pragma unroll
  for (int j = 0; j < 8; j++) {
    float v = (float)x[(long)b * 256 + lane + j * 32];
    vals[j] = v; ss += v * v;
  }
#pragma unroll
  for (int o = 16; o > 0; o >>= 1) ss += __shfl_xor(ss, o, 32);
  float s = -(1.f - ss) / (1.f + ss);
#pragma unroll
  for (int j = 0; j < 8; j++) out[(long)b * 257 + lane + j * 32] = (1.f - s) * vals[j];
  if (lane == 0) out[(long)b * 257 + 256] = s;
}

// ---------------- host driver ----------------

extern "C" void kernel_launch(void* const* d_in, const int* in_sizes, int n_in,
                              void* d_out, int out_size, void* d_ws, size_t ws_size,
                              hipStream_t stream) {
  (void)in_sizes; (void)n_in; (void)out_size; (void)ws_size;

  int ii = 0;
  auto F = [&]() { return (const float*)d_in[ii++]; };
  struct BNp { const float *g, *b, *m, *v; };
  struct Blk { const float* w1; BNp b1; const float* w2; BNp b2; const float* w3; BNp b3;
               const float* dw; BNp db; };
  auto getBN = [&]() { BNp p; p.g = F(); p.b = F(); p.m = F(); p.v = F(); return p; };
  auto getBlk = [&](bool down) {
    Blk k{}; k.w1 = F(); k.b1 = getBN(); k.w2 = F(); k.b2 = getBN();
    k.w3 = F(); k.b3 = getBN();
    if (down) { k.dw = F(); k.db = getBN(); }
    return k;
  };

  const float* X = F();            // [2048,3,32,32]
  const float* conv1_w = F();      // [64,3,7,7]
  BNp bn1 = getBN();
  Blk blks[7];
  blks[0] = getBlk(true);  blks[1] = getBlk(false); blks[2] = getBlk(false);
  blks[3] = getBlk(true);  blks[4] = getBlk(false); blks[5] = getBlk(false); blks[6] = getBlk(false);
  const float* fc_w  = F(); const float* fc_b  = F();   // [512,8192],[512]
  const float* fc1_w = F(); const float* fc1_b = F();   // [256,512],[256]

  // bump allocator over d_ws
  size_t off = 0;
  auto alloc = [&](size_t bytes) -> void* {
    off = (off + 255) & ~(size_t)255;
    void* p = (char*)d_ws + off;
    off += bytes;
    return p;
  };

  const int B = 2048;
  _Float16* buf0 = (_Float16*)alloc((size_t)B * 16 * 16 * 64 * 2);  // 67 MB
  _Float16* buf1 = (_Float16*)alloc((size_t)B * 16 * 16 * 64 * 2);  // 67 MB
  _Float16* mid0 = (_Float16*)alloc((size_t)B * 8 * 8 * 128 * 2);   // 33.5 MB
  _Float16* mid1 = (_Float16*)alloc((size_t)B * 8 * 8 * 128 * 2);   // 33.5 MB

  struct Cfg { int cin, mid, cout, isp, osp, stride; bool down; };
  const Cfg cfg[7] = {
    {64,  64, 256, 8, 8, 1, true }, {256, 64, 256, 8, 8, 1, false}, {256, 64, 256, 8, 8, 1, false},
    {256,128, 512, 8, 4, 2, true }, {512,128, 512, 4, 4, 1, false}, {512,128, 512, 4, 4, 1, false},
    {512,128, 512, 4, 4, 1, false}};

  auto prepw = [&](const float* w, int O, int Cin, int KH) {
    int Kpad = ((Cin * KH * KH) + 31) / 32 * 32;
    _Float16* wt = (_Float16*)alloc((size_t)O * Kpad * 2);
    long tot = (long)O * Kpad;
    prep_w_kernel<<<(unsigned)((tot + 255) / 256), 256, 0, stream>>>(w, wt, O, Cin, KH, Kpad);
    return wt;
  };
  auto fold = [&](const BNp& p, int C, float** sc, float** bi) {
    *sc = (float*)alloc((size_t)C * 4); *bi = (float*)alloc((size_t)C * 4);
    bn_fold_kernel<<<(C + 255) / 256, 256, 0, stream>>>(p.g, p.b, p.m, p.v, *sc, *bi, C);
  };

  _Float16* wt_c1 = prepw(conv1_w, 64, 3, 7);   // Kpad 160
  float *sc_c1, *bi_c1; fold(bn1, 64, &sc_c1, &bi_c1);

  _Float16 *wt1[7], *wt2[7], *wt3[7], *wtd[7];
  float *s1[7], *b1a[7], *s2[7], *b2a[7], *s3[7], *b3a[7], *sd[7], *bd[7];
  for (int i = 0; i < 7; i++) {
    const Cfg& c = cfg[i];
    wt1[i] = prepw(blks[i].w1, c.mid,  c.cin, 1);
    wt2[i] = prepw(blks[i].w2, c.mid,  c.mid, 3);
    wt3[i] = prepw(blks[i].w3, c.cout, c.mid, 1);
    fold(blks[i].b1, c.mid,  &s1[i], &b1a[i]);
    fold(blks[i].b2, c.mid,  &s2[i], &b2a[i]);
    fold(blks[i].b3, c.cout, &s3[i], &b3a[i]);
    wtd[i] = nullptr; sd[i] = nullptr; bd[i] = nullptr;
    if (c.down) { wtd[i] = prepw(blks[i].dw, c.cout, c.cin, 1); fold(blks[i].db, c.cout, &sd[i], &bd[i]); }
  }
  _Float16* wt_fc = (_Float16*)alloc((size_t)512 * 8192 * 2);
  prep_fc_kernel<<<(512 * 8192) / 256, 256, 0, stream>>>(fc_w, wt_fc);
  _Float16* wt_fc1 = prepw(fc1_w, 256, 512, 1);

  auto convg = [&](const _Float16* in, const _Float16* wt, const float* sc, const float* bi,
                   const _Float16* res, _Float16* out, int Cin, int Cout, int IH, int OH,
                   int KH, int stride, int pad, int relu) {
    long M = (long)B * OH * OH;
    int Kpad = ((Cin * KH * KH) + 31) / 32 * 32;
    dim3 g((unsigned)(M / 128), (unsigned)(Cout / 64));
    convg_kernel<<<g, 256, 0, stream>>>(in, wt, sc, bi, res, out, Cin, Cout, IH, OH, KH,
                                        stride, pad, Kpad, relu);
  };
  auto gemm = [&](const _Float16* in, const _Float16* wt, const float* sc, const float* bi,
                  const _Float16* res, _Float16* out, int OW, int IW, int stride,
                  int Cin, int Cout, int relu) {
    long M = (long)B * OW * OW;
    dim3 g((unsigned)(M / 128), (unsigned)(Cout / 64));
    gemm1x1_kernel<<<g, 256, 0, stream>>>(in, wt, sc, bi, res, out, OW, IW, stride, Cin, Cout, relu);
  };

  // ---- forward ----
  _Float16* xn = mid0;   // NHWC f16 input (6.3M halfs, fits in mid0)
  {
    long tot = (long)B * 3 * 32 * 32;
    nchw2nhwc_kernel<<<(unsigned)((tot + 255) / 256), 256, 0, stream>>>(X, xn, B, 3, 32, 32);
  }
  convg(xn, wt_c1, sc_c1, bi_c1, nullptr, buf0, 3, 64, 32, 16, 7, 2, 3, 1);  // conv1+bn+relu
  {
    long tot = (long)B * 8 * 8 * 64;
    maxpool_kernel<<<(unsigned)((tot + 255) / 256), 256, 0, stream>>>(buf0, buf1, B, 64, 16, 8);
  }

  _Float16* cur = buf1; _Float16* alt = buf0;
  for (int i = 0; i < 7; i++) {
    const Cfg& c = cfg[i];
    // t1 = relu(bn(1x1 conv))
    gemm(cur, wt1[i], s1[i], b1a[i], nullptr, mid0, c.isp, c.isp, 1, c.cin, c.mid, 1);
    // t2 = relu(bn(3x3 conv, stride))
    convg(mid0, wt2[i], s2[i], b2a[i], nullptr, mid1, c.mid, c.mid, c.isp, c.osp, 3, c.stride, 1, 1);
    const _Float16* res; _Float16* ob = alt;
    if (c.down) {  // idn = bn(1x1 stride conv); then w3 adds residual in-place
      gemm(cur, wtd[i], sd[i], bd[i], nullptr, alt, c.osp, c.isp, c.stride, c.cin, c.cout, 0);
      res = alt;
    } else {
      res = cur;
    }
    // out = relu(bn(1x1 conv) + idn)
    gemm(mid1, wt3[i], s3[i], b3a[i], res, ob, c.osp, c.osp, 1, c.mid, c.cout, 1);
    _Float16* t = cur; cur = ob; alt = t;
  }

  // FC: NHWC [B,4,4,512] flattens to K=8192 matching the permuted fc weights
  _Float16* fco  = mid0;
  gemm(cur, wt_fc,  nullptr, fc_b,  nullptr, fco,  1, 1, 1, 8192, 512, 1);
  _Float16* fc1o = mid1;
  gemm(fco, wt_fc1, nullptr, fc1_b, nullptr, fc1o, 1, 1, 1, 512,  256, 1);

  project_kernel<<<B / 8, 256, 0, stream>>>(fc1o, (float*)d_out, B);
}
